// MultiHeadedAttention_5858335392474
// MI455X (gfx1250) — matrix operand order repaired and checked by reading, not compiled, not run on hardware
//
#include <hip/hip_runtime.h>
#include <hip/hip_bf16.h>

// ---------------------------------------------------------------------------
// MultiHeadedAttention for MI455X (gfx1250, wave32, WMMA bf16 path)
// B=2, S=2048, D=1024, H=16, DH=64
// GEMMs use async global->LDS staging (ASYNCcnt) with double buffering.
// ---------------------------------------------------------------------------

#define DEV __device__ __forceinline__

typedef __attribute__((ext_vector_type(16))) __bf16          bf16x16;
typedef __attribute__((ext_vector_type(8)))  float           f32x8;
typedef __attribute__((ext_vector_type(8)))  unsigned short  u16x8;
typedef __attribute__((ext_vector_type(16))) unsigned short  u16x16;

static constexpr int kB  = 2;
static constexpr int kS  = 2048;
static constexpr int kD  = 1024;
static constexpr int kH  = 16;
static constexpr int kDH = 64;
static constexpr int kQKVLD = 3 * kD;   // 3072 row stride of fused qkv
static constexpr int kRows  = kB * kS;  // 4096

static constexpr int kLdsStride = 40;   // 32 k-elems + 8 pad (bank-conflict-free)

// float -> bf16, round-to-nearest-even (fast path, NaN-agnostic)
static DEV unsigned short f2bf(float f) {
    union { float f; unsigned u; } x; x.f = f;
    unsigned r = x.u + 0x7FFFu + ((x.u >> 16) & 1u);
    return (unsigned short)(r >> 16);
}

// A-fragment (16x32 bf16, M = lane%16): 16 contiguous elements per
// lane at k-offset koff, where koff = (lane>=16 ? 16 : 0).
static DEV bf16x16 load_a_frag(const unsigned short* p, int koff) {
    union { u16x16 v; u16x8 h[2]; } u;
    u.h[0] = *(const u16x8*)(p + koff);
    u.h[1] = *(const u16x8*)(p + koff + 8);
    return __builtin_bit_cast(bf16x16, u.v);
}

// B-fragment (32x16 bf16, N = lane%16): 16 contiguous bf16 per lane, caller
// has already folded in the (lane>=16 ? 16 : 0) k-offset. Two 16B loads so
// only 16-byte alignment is required (works with padded LDS strides).
static DEV bf16x16 load_b_frag(const unsigned short* p) {
    union { u16x16 v; u16x8 h[2]; } u;
    u.h[0] = *(const u16x8*)(p);
    u.h[1] = *(const u16x8*)(p + 8);
    return __builtin_bit_cast(bf16x16, u.v);
}

static DEV f32x8 wmma_bf16(bf16x16 a, bf16x16 b, f32x8 c) {
    return __builtin_amdgcn_wmma_f32_16x16x32_bf16(
        /*neg_a=*/false, a, /*neg_b=*/false, b,
        /*c_mod=*/(short)0, c, /*reuse_a=*/false, /*reuse_b=*/false);
}

// one async 16B global->LDS chunk (per-lane addresses, tracked by ASYNCcnt)
static DEV void async_b128_to_lds(unsigned lds_byte_addr, const unsigned short* g) {
    asm volatile("global_load_async_to_lds_b128 %0, %1, off"
                 :: "v"(lds_byte_addr),
                    "v"((unsigned long long)(uintptr_t)g)
                 : "memory");
}

static DEV void wait_async_zero() {
    asm volatile("s_wait_asynccnt 0x0" ::: "memory");
}

// Stage a [128 rows x 32 k] bf16 tile into LDS (padded stride kLdsStride).
// 512 16B chunks, 256 threads -> 2 async instructions per wave per tile.
static DEV void stage_tile_async(unsigned lds_base, const unsigned short* g_base,
                                 int ldg, int tid) {
#pragma unroll
    for (int s = 0; s < 2; ++s) {
        const int c   = s * 256 + tid;   // 0..511
        const int row = c >> 2;          // 0..127
        const int kc  = (c & 3) * 8;     // 0,8,16,24
        async_b128_to_lds(lds_base + (unsigned)(row * kLdsStride + kc) * 2u,
                          g_base + (size_t)row * ldg + kc);
    }
}

// ---------------------------------------------------------------------------
// fp32 -> bf16 conversion
// ---------------------------------------------------------------------------
__global__ void cvt_f32_to_bf16(const float* __restrict__ in,
                                unsigned short* __restrict__ out, long n) {
    long i = (long)blockIdx.x * blockDim.x + threadIdx.x;
    if (i < n) out[i] = f2bf(in[i]);
}

// ---------------------------------------------------------------------------
// GEMM: C[M,N] = A[M,K] * W[N,K]^T (+bias).  A, W row-major bf16.
// 256 threads = 8 waves; wave tile 32x64; block tile 128x128.
// A/B tiles double-buffered in LDS via async global->LDS copies.
// ---------------------------------------------------------------------------
template <bool OUT_BF16>
__global__ void __launch_bounds__(256)
gemm_bf16_nt(const unsigned short* __restrict__ A,
             const unsigned short* __restrict__ W,
             const float* __restrict__ bias,
             float* __restrict__ Cf,
             unsigned short* __restrict__ Cb,
             int M, int N, int K) {
    __shared__ alignas(16) unsigned short sA[2][128 * kLdsStride];  // 20 KB
    __shared__ alignas(16) unsigned short sB[2][128 * kLdsStride];  // 20 KB

    const int tid  = threadIdx.x;
    const int lane = tid & 31;
    const int wave = tid >> 5;
    const int hh   = lane >> 4;   // lane half (0/1)
    const int l16  = lane & 15;
    const int wm   = wave & 3;    // 4 waves along M
    const int wn   = wave >> 2;   // 2 waves along N
    const int mblk = blockIdx.y * 128;
    const int nblk = blockIdx.x * 128;

    const unsigned ldsA[2] = { (unsigned)(uintptr_t)&sA[0][0],
                               (unsigned)(uintptr_t)&sA[1][0] };
    const unsigned ldsB[2] = { (unsigned)(uintptr_t)&sB[0][0],
                               (unsigned)(uintptr_t)&sB[1][0] };

    f32x8 acc[2][4];
#pragma unroll
    for (int i = 0; i < 2; ++i)
#pragma unroll
        for (int j = 0; j < 4; ++j) acc[i][j] = {};

    const int koff = hh ? 16 : 0;
    const int nk = K / 32;

    // prologue: stage k-slice 0 into buffer 0
    stage_tile_async(ldsA[0], A + (size_t)mblk * K, K, tid);
    stage_tile_async(ldsB[0], W + (size_t)nblk * K, K, tid);

    for (int ik = 0; ik < nk; ++ik) {
        const int buf = ik & 1;

        // stage ik copies are the only outstanding async ops
        wait_async_zero();
        __syncthreads();  // writes of all waves visible; prior reads of buf^1 done

        if (ik + 1 < nk) {  // overlap next copy with this stage's WMMAs
            const int kb = (ik + 1) * 32;
            stage_tile_async(ldsA[buf ^ 1], A + (size_t)mblk * K + kb, K, tid);
            stage_tile_async(ldsB[buf ^ 1], W + (size_t)nblk * K + kb, K, tid);
        }

        const unsigned short* aT = &sA[buf][0];
        const unsigned short* bT = &sB[buf][0];
        bf16x16 af[2];
#pragma unroll
        for (int mt = 0; mt < 2; ++mt)
            af[mt] = load_a_frag(aT + (wm * 32 + mt * 16 + l16) * kLdsStride, koff);
#pragma unroll
        for (int t = 0; t < 4; ++t) {
            bf16x16 b = load_b_frag(
                bT + (wn * 64 + t * 16 + l16) * kLdsStride + (hh ? 16 : 0));
            acc[0][t] = wmma_bf16(af[0], b, acc[0][t]);
            acc[1][t] = wmma_bf16(af[1], b, acc[1][t]);
        }
    }

#pragma unroll
    for (int mt = 0; mt < 2; ++mt) {
#pragma unroll
        for (int t = 0; t < 4; ++t) {
            const int col = nblk + wn * 64 + t * 16 + l16;
            const float bv = bias ? bias[col] : 0.0f;
#pragma unroll
            for (int v = 0; v < 8; ++v) {
                const int row = mblk + wm * 32 + mt * 16 + (hh ? 8 : 0) + v;
                float x = acc[mt][t][v] + bv;
                if (OUT_BF16)
                    Cb[(size_t)row * N + col] = f2bf(x);
                else
                    Cf[(size_t)row * N + col] = x;
            }
        }
    }
}

// ---------------------------------------------------------------------------
// Flash-style attention. Grid (S/64, H, B); block = 128 threads = 4 waves.
// Each wave owns 16 query rows; iterates key blocks of 64 with online softmax.
// qkv: bf16 [4096, 3072] (Q|K|V), mask: int [B,S,S], O: bf16 [4096, 1024].
// ---------------------------------------------------------------------------
__global__ void __launch_bounds__(128)
attn_fwd(const unsigned short* __restrict__ qkv,
         const int* __restrict__ mask,
         unsigned short* __restrict__ O) {
    __shared__ alignas(32) unsigned short Vt[kDH][64];     // V^T block [dh][key]
    __shared__ alignas(32) unsigned short Pb[4][16][64];   // probs per wave

    const int b  = blockIdx.z;
    const int h  = blockIdx.y;
    const int qb = blockIdx.x;
    const int lane = threadIdx.x & 31;
    const int wave = threadIdx.x >> 5;
    const int hh   = lane >> 4;
    const int l16  = lane & 15;
    const int koff = hh ? 16 : 0;

    const int qbase = qb * 64 + wave * 16;

    // Q A-fragments (held in registers for the whole kernel)
    const unsigned short* qp =
        qkv + ((size_t)b * kS + qbase + l16) * kQKVLD + h * kDH;
    bf16x16 aq0 = load_a_frag(qp, koff);        // d = 0..31
    bf16x16 aq1 = load_a_frag(qp + 32, koff);   // d = 32..63

    const unsigned short* kcol =
        qkv + (size_t)b * kS * kQKVLD + kD + h * kDH + (hh ? 16 : 0);
    const unsigned short* vcol =
        qkv + (size_t)b * kS * kQKVLD + 2 * kD + h * kDH;

    f32x8 accO[4];
#pragma unroll
    for (int t = 0; t < 4; ++t) accO[t] = {};
    f32x8 mrun, lrun;
#pragma unroll
    for (int v = 0; v < 8; ++v) { mrun[v] = -3.0e38f; lrun[v] = 0.0f; }

    const float scale = 0.125f;  // 1/sqrt(64)

    for (int jb = 0; jb < kS; jb += 64) {
        // ---- stage V^T into LDS (cooperative, 128 threads) ----
        __syncthreads();  // protect Vt from previous iteration readers
        {
            const int j  = threadIdx.x >> 1;
            const int d0 = (threadIdx.x & 1) * 32;
            const unsigned short* vp = vcol + (size_t)(jb + j) * kQKVLD + d0;
#pragma unroll
            for (int i = 0; i < 32; i += 8) {
                u16x8 vv = *(const u16x8*)(vp + i);
#pragma unroll
                for (int e = 0; e < 8; ++e) Vt[d0 + i + e][j] = vv[e];
            }
        }
        __syncthreads();

        // ---- scores: S = Q * K^T (K rows are contiguous -> B-fragments) ----
        f32x8 sc[4];
#pragma unroll
        for (int t = 0; t < 4; ++t) {
            const unsigned short* kp = kcol + (size_t)(jb + t * 16 + l16) * kQKVLD;
            f32x8 c = {};
            c = wmma_bf16(aq0, load_b_frag(kp), c);
            c = wmma_bf16(aq1, load_b_frag(kp + 32), c);
            sc[t] = c;
        }

        // ---- scale + mask ----
#pragma unroll
        for (int t = 0; t < 4; ++t) {
            const int col = jb + t * 16 + l16;
#pragma unroll
            for (int v = 0; v < 8; ++v) {
                const int qrow = qbase + (hh ? 8 : 0) + v;
                const int mv = mask[((size_t)b * kS + qrow) * kS + col];
                const float s = sc[t][v] * scale;
                sc[t][v] = (mv == 0) ? -1e9f : s;
            }
        }

        // ---- online softmax (rows live in 16-lane halves) ----
        f32x8 rmax = sc[0];
#pragma unroll
        for (int t = 1; t < 4; ++t)
#pragma unroll
            for (int v = 0; v < 8; ++v) rmax[v] = fmaxf(rmax[v], sc[t][v]);
#pragma unroll
        for (int m = 1; m < 16; m <<= 1)
#pragma unroll
            for (int v = 0; v < 8; ++v)
                rmax[v] = fmaxf(rmax[v], __shfl_xor(rmax[v], m, 32));

        f32x8 nm, corr;
#pragma unroll
        for (int v = 0; v < 8; ++v) {
            nm[v]   = fmaxf(mrun[v], rmax[v]);
            corr[v] = __expf(mrun[v] - nm[v]);
        }
        f32x8 rsum = {};
#pragma unroll
        for (int t = 0; t < 4; ++t)
#pragma unroll
            for (int v = 0; v < 8; ++v) {
                const float p = __expf(sc[t][v] - nm[v]);
                sc[t][v] = p;
                rsum[v] += p;
            }
#pragma unroll
        for (int m = 1; m < 16; m <<= 1)
#pragma unroll
            for (int v = 0; v < 8; ++v)
                rsum[v] += __shfl_xor(rsum[v], m, 32);
#pragma unroll
        for (int v = 0; v < 8; ++v) {
            lrun[v] = lrun[v] * corr[v] + rsum[v];
            mrun[v] = nm[v];
        }
#pragma unroll
        for (int t = 0; t < 4; ++t)
#pragma unroll
            for (int v = 0; v < 8; ++v) accO[t][v] *= corr[v];

        // ---- P -> LDS (bf16, wave-private; DS ops are in-order per wave) ----
#pragma unroll
        for (int t = 0; t < 4; ++t)
#pragma unroll
            for (int v = 0; v < 8; ++v)
                Pb[wave][(hh ? 8 : 0) + v][t * 16 + l16] = f2bf(sc[t][v]);

        // ---- O += P * V  (A from Pb rows, B from Vt rows, all contiguous) --
#pragma unroll
        for (int c2 = 0; c2 < 2; ++c2) {
            bf16x16 ap = load_a_frag(&Pb[wave][l16][c2 * 32], koff);
#pragma unroll
            for (int t = 0; t < 4; ++t) {
                bf16x16 bv =
                    load_b_frag(&Vt[t * 16 + l16][c2 * 32 + (hh ? 16 : 0)]);
                accO[t] = wmma_bf16(ap, bv, accO[t]);
            }
        }
    }

    // ---- epilogue: O = accO / l, store bf16 in heads-merged layout ----
    f32x8 inv;
#pragma unroll
    for (int v = 0; v < 8; ++v) {
        const float l = lrun[v];
        inv[v] = (l > 0.0f) ? (1.0f / l) : 0.0f;
    }
#pragma unroll
    for (int t = 0; t < 4; ++t) {
        const int col = h * kDH + t * 16 + l16;
#pragma unroll
        for (int v = 0; v < 8; ++v) {
            const int row = b * kS + qbase + (hh ? 8 : 0) + v;
            O[(size_t)row * kD + col] = f2bf(accO[t][v] * inv[v]);
        }
    }
}

// ---------------------------------------------------------------------------
// Host-side orchestration (graph-capture safe: only kernel launches)
// ---------------------------------------------------------------------------
extern "C" void kernel_launch(void* const* d_in, const int* in_sizes, int n_in,
                              void* d_out, int out_size, void* d_ws, size_t ws_size,
                              hipStream_t stream) {
    const float* X     = (const float*)d_in[0];  // [B,S,D]
    const int*   mask  = (const int*)d_in[1];    // [B,S,S]
    const float* Wqkv  = (const float*)d_in[2];  // [3D,D]
    const float* Wout  = (const float*)d_in[3];  // [D,D]
    const float* bout  = (const float*)d_in[4];  // [D]
    float*       out   = (float*)d_out;          // [B,S,D]

    // workspace carve (all offsets 256B-aligned)
    char* ws = (char*)d_ws;
    unsigned short* Xb    = (unsigned short*)(ws);                 //  8 MB
    unsigned short* Wqkvb = (unsigned short*)(ws + 8388608);       //  6 MB
    unsigned short* Woutb = (unsigned short*)(ws + 14680064);      //  2 MB
    unsigned short* QKVb  = (unsigned short*)(ws + 16777216);      // 24 MB
    unsigned short* Ob    = (unsigned short*)(ws + 41943040);      //  8 MB

    const long nX  = (long)kRows * kD;        // 4,194,304
    const long nWq = (long)3 * kD * kD;       // 3,145,728
    const long nWo = (long)kD * kD;           // 1,048,576

    cvt_f32_to_bf16<<<(nX  + 255) / 256, 256, 0, stream>>>(X,    Xb,    nX);
    cvt_f32_to_bf16<<<(nWq + 255) / 256, 256, 0, stream>>>(Wqkv, Wqkvb, nWq);
    cvt_f32_to_bf16<<<(nWo + 255) / 256, 256, 0, stream>>>(Wout, Woutb, nWo);

    // QKV = X @ Wqkv^T : [4096,1024] x [3072,1024]^T -> bf16 [4096,3072]
    gemm_bf16_nt<true><<<dim3(3 * kD / 128, kRows / 128), 256, 0, stream>>>(
        Xb, Wqkvb, nullptr, nullptr, QKVb, kRows, 3 * kD, kD);

    // attention -> Ob bf16 [4096,1024]
    attn_fwd<<<dim3(kS / 64, kH, kB), 128, 0, stream>>>(QKVb, mask, Ob);

    // out = Ob @ Wout^T + b_out : f32 [4096,1024]
    gemm_bf16_nt<false><<<dim3(kD / 128, kRows / 128), 256, 0, stream>>>(
        Ob, Woutb, bout, out, nullptr, kRows, kD, kD);
}
